// MultiHeadAttention_48481590837801
// MI455X (gfx1250) — matrix operand order repaired and checked by reading, not compiled, not run on hardware
//
#include <hip/hip_runtime.h>

// ---------------------------------------------------------------------------
// MultiHeadAttention + RoPE for MI455X (gfx1250), wave32, WMMA bf16 pipeline.
// B=2, S=2048, D=1024, H=16, Dk=64.  Compute-bound => bf16 WMMA, fp32 accum.
// Projection GEMMs: 64x64 workgroup tiles, weight tiles staged into LDS with
// the Tensor Data Mover (tensor_load_to_lds), double buffered on TENSORcnt,
// A-stream covered with global_prefetch.
// ---------------------------------------------------------------------------

#define D_MODEL   1024
#define NUM_HEADS 16
#define D_K       64
#define SEQ       2048
#define BATCH     2

typedef __attribute__((ext_vector_type(16))) __bf16       v16bf;
typedef __attribute__((ext_vector_type(8)))  float        v8f;
typedef __attribute__((ext_vector_type(4)))  unsigned int u32x4;
typedef __attribute__((ext_vector_type(8)))  unsigned int u32x8;

#define WMMA_BF16(A, B, C) \
    __builtin_amdgcn_wmma_f32_16x16x32_bf16(false, (A), false, (B), (short)0, (C), false, false)

#define WAIT_TENSORCNT_0() asm volatile("s_wait_tensorcnt 0x0" ::: "memory")
#define WAIT_TENSORCNT_1() asm volatile("s_wait_tensorcnt 0x1" ::: "memory")

// ---------------------------------------------------------------------------
// TDM: DMA a 64-row x 32-col bf16 tile (row stride 1024 elems) into LDS.
// D# per CDNA5 ISA 8.3/8.4: group0 = {count=1, lds_addr, gaddr57, type=2};
// group1 = {data_size=2B, tensor_dim0/1=1024, tile=32x64, stride0=1024}.
// All computed words are pushed through readfirstlane so the descriptor is
// provably uniform and binds to SGPR tuples (TDM requires SGPR operands).
// ---------------------------------------------------------------------------
__device__ __forceinline__ void tdm_load_w_tile(const __bf16* gsrc, __bf16* lds_dst) {
    unsigned long long ga = (unsigned long long)(uintptr_t)gsrc;
    unsigned lof  = __builtin_amdgcn_readfirstlane((unsigned)(uintptr_t)lds_dst);
    unsigned galo = __builtin_amdgcn_readfirstlane((unsigned)(ga & 0xffffffffu));
    unsigned gahi = __builtin_amdgcn_readfirstlane((unsigned)((ga >> 32) & 0x1ffffffu));
    u32x4 g0;
    g0[0] = 1u;                                          // count=1, user mode
    g0[1] = lof;                                         // lds_addr (byte offset)
    g0[2] = galo;                                        // global_addr[31:0]
    g0[3] = gahi | (2u << 30);                           // global_addr[56:32] | type=2
    u32x8 g1;
    g1[0] = 1u << 16;                                    // data_size=1 (2 bytes)
    g1[1] = (1024u & 0xffffu) << 16;                     // tensor_dim0 lo16
    g1[2] = (1024u >> 16) | ((1024u & 0xffffu) << 16);   // td0 hi | tensor_dim1 lo
    g1[3] = (1024u >> 16) | (32u << 16);                 // td1 hi | tile_dim0=32
    g1[4] = 64u;                                         // tile_dim1=64, tile_dim2=0
    g1[5] = 1024u;                                       // tensor_dim0_stride lo32
    g1[6] = 0u;                                          // stride0 hi | stride1 lo
    g1[7] = 0u;                                          // stride1 hi
    asm volatile("tensor_load_to_lds %0, %1" :: "s"(g0), "s"(g1) : "memory");
}

// ---------------------------------------------------------------------------
// fp32 -> bf16 elementwise convert
// ---------------------------------------------------------------------------
__global__ void cvt_f32_to_bf16(const float* __restrict__ src,
                                __bf16* __restrict__ dst, int n) {
    int i = blockIdx.x * blockDim.x + threadIdx.x;
    if (i < n) dst[i] = (__bf16)src[i];
}

// A-fragment (16x32 bf16): lane (r=lane&15,hi=lane>>4) holds row r,
// k = hi*16+{0..15}  -> two contiguous 16B loads.
__device__ __forceinline__ v16bf load_a32(const __bf16* row, int hi) {
    union { v16bf v; uint4 q[2]; } u;
    u.q[0] = *(const uint4*)(row + hi * 16);
    u.q[1] = *(const uint4*)(row + hi * 16 + 8);
    return u.v;
}

// Shared GEMM core: 4 waves, 64 rows x 64 cols per workgroup, W tile staged
// in LDS by the TDM, double buffered.  Each wave holds 4 named accumulators
// (in-place WMMA accumulation, avoids accumulator copy/NOP hazards).
// Blds layout: row o_local (0..63) at offset o_local*32 elements (k-major).
__device__ __forceinline__ void gemm64x64_tdm(const __bf16* __restrict__ arow,
                                              const __bf16* __restrict__ wtile,
                                              __bf16 (&Blds)[2][64 * 32],
                                              int wave, int r, int hi,
                                              v8f& c0, v8f& c1, v8f& c2, v8f& c3) {
    const int NK = D_MODEL / 32;
    if (wave == 0) tdm_load_w_tile(wtile, &Blds[0][0]);
    for (int ks = 0; ks < NK; ++ks) {
        __syncthreads();                       // prev reads of next buf done
        if (wave == 0) {
            if (ks + 1 < NK) {
                tdm_load_w_tile(wtile + (ks + 1) * 32, &Blds[(ks + 1) & 1][0]);
                WAIT_TENSORCNT_1();            // oldest (this ks) complete
            } else {
                WAIT_TENSORCNT_0();
            }
        }
        if (ks + 1 < NK)                       // cover A-stream latency
            __builtin_prefetch(arow + (ks + 1) * 32, 0, 0);
        __syncthreads();                       // tile for ks visible to all
        v16bf a = load_a32(arow + ks * 32, hi);
        const __bf16* bl = &Blds[ks & 1][0] + r * 32 + hi * 16;
        v16bf b0 = *(const v16bf*)(bl);
        v16bf b1 = *(const v16bf*)(bl + 16 * 32);
        v16bf b2 = *(const v16bf*)(bl + 32 * 32);
        v16bf b3 = *(const v16bf*)(bl + 48 * 32);
        c0 = WMMA_BF16(a, b0, c0);
        c1 = WMMA_BF16(a, b1, c1);
        c2 = WMMA_BF16(a, b2, c2);
        c3 = WMMA_BF16(a, b3, c3);
    }
}

// ---------------------------------------------------------------------------
// Fused QKV projection + RoPE.  grid = (64 M64tiles, 16 heads, 3), block 128.
// z=0 -> Q (rope), z=1 -> K (rope), z=2 -> V (packed B-fragment layout).
// N64 tile == one head (D_K=64).
// ---------------------------------------------------------------------------
__global__ void qkv_gemm_rope(const __bf16* __restrict__ xb,
                              const __bf16* __restrict__ wb,
                              __bf16* __restrict__ Qr,
                              __bf16* __restrict__ Kr,
                              __bf16* __restrict__ Vp) {
    __shared__ __align__(64) __bf16 Blds[2][64 * 32];

    const int lane = threadIdx.x & 31;
    const int wave = __builtin_amdgcn_readfirstlane(threadIdx.x >> 5);
    const int r  = lane & 15;
    const int hi = lane >> 4;
    const int mt64 = blockIdx.x;
    const int nt   = blockIdx.y;           // head index (64-col tile)
    const int z    = blockIdx.z;

    const __bf16* W     = wb + (size_t)z * (D_MODEL * (size_t)D_MODEL);
    const __bf16* wtile = W + (size_t)(nt * 64) * D_MODEL;          // 64 rows of W
    const int     mt16  = mt64 * 4 + wave;                          // 16-row tile id
    const __bf16* arow  = xb + (size_t)(mt16 * 16 + r) * D_MODEL;

    v8f c0, c1, c2, c3;
#pragma unroll
    for (int i = 0; i < 8; ++i) { c0[i] = 0.0f; c1[i] = 0.0f; c2[i] = 0.0f; c3[i] = 0.0f; }

    gemm64x64_tdm(arow, wtile, Blds, wave, r, hi, c0, c1, c2, c3);

    v8f cacc[4] = { c0, c1, c2, c3 };
    const int b    = mt16 >> 7;
    const int sbas = (mt16 & 127) * 16;
    if (z < 2) {
        // ---- RoPE + row-major store [b][h][s][64] ----
        __bf16* dst = (z == 0 ? Qr : Kr);
#pragma unroll
        for (int dq = 0; dq < 4; ++dq) {
            const int dl = dq * 16 + r;                  // dim within head
            const float invf = __expf(-(float)(dl & ~1) * 0.14391156f); // ln(1e4)/64
#pragma unroll
            for (int e = 0; e < 8; ++e) {
                int   m    = hi * 8 + e;
                int   spos = sbas + m;
                float a    = cacc[dq][e];
                float prt  = __shfl_xor(a, 1);           // even<->odd partner
                float sn, cs;
                __sincosf((float)spos * invf, &sn, &cs);
                float out = (dl & 1) ? (prt * sn + a * cs)
                                     : (a * cs - prt * sn);
                dst[(((size_t)(b * NUM_HEADS + nt) * SEQ + spos) * D_K) + dl] = (__bf16)out;
            }
        }
    } else {
        // ---- V: pack into B-fragment layout [b][h][s32][dq][lane32][16] ----
        const int st32  = (mt16 & 127) >> 1;
        const int khalf = mt16 & 1;
        const int j     = khalf * 16 + r;
#pragma unroll
        for (int dq = 0; dq < 4; ++dq) {
            size_t base = ((((size_t)(b * NUM_HEADS + nt) * (SEQ / 32) + st32) * 4 + dq) * 32 + j)
                              * 16 + hi * 8;
            union { __bf16 h8[8]; uint4 u; } pk;
#pragma unroll
            for (int e = 0; e < 8; ++e) pk.h8[e] = (__bf16)cacc[dq][e];
            *(uint4*)(Vp + base) = pk.u;                 // contiguous 16B store
        }
    }
}

// ---------------------------------------------------------------------------
// Flash attention, causal.  grid = (S/16, H, B), block = 32 (one wave).
// ---------------------------------------------------------------------------
__global__ void attn_kernel(const __bf16* __restrict__ Q,
                            const __bf16* __restrict__ K,
                            const __bf16* __restrict__ Vp,
                            __bf16* __restrict__ Ob) {
    __shared__ __align__(32) __bf16 Plds[16 * 32];

    const int lane = threadIdx.x;
    const int r  = lane & 15;
    const int hi = lane >> 4;
    const int it = blockIdx.x;
    const int h  = blockIdx.y;
    const int b  = blockIdx.z;

    const __bf16* qrow = Q + ((size_t)(b * NUM_HEADS + h) * SEQ + it * 16 + r) * D_K;
    v16bf aq0 = load_a32(qrow, hi);
    v16bf aq1 = load_a32(qrow + 32, hi);

    float mi[8], li[8];
    v8f   oacc[4];
#pragma unroll
    for (int e = 0; e < 8; ++e) { mi[e] = -1e30f; li[e] = 0.0f; }
#pragma unroll
    for (int dq = 0; dq < 4; ++dq)
#pragma unroll
        for (int e = 0; e < 8; ++e) oacc[dq][e] = 0.0f;

    const float scale = 0.125f;
    const int   jmax  = (it * 16 + 15) >> 5;
    const __bf16* kbase = K + (size_t)(b * NUM_HEADS + h) * SEQ * D_K;
    const v16bf*  vbase = (const v16bf*)Vp
                          + (size_t)(b * NUM_HEADS + h) * (SEQ / 32) * (4 * 32);

    for (int jt = 0; jt <= jmax; ++jt) {
        const __bf16* k0 = kbase + (size_t)(jt * 32 + r) * D_K + hi * 16;
        const __bf16* k1 = k0 + 16 * D_K;
        v16bf bk0a = *(const v16bf*)(k0);
        v16bf bk0b = *(const v16bf*)(k0 + 32);
        v16bf bk1a = *(const v16bf*)(k1);
        v16bf bk1b = *(const v16bf*)(k1 + 32);

        v8f c0, c1;
#pragma unroll
        for (int i = 0; i < 8; ++i) { c0[i] = 0.0f; c1[i] = 0.0f; }
        c0 = WMMA_BF16(aq0, bk0a, c0);
        c0 = WMMA_BF16(aq1, bk0b, c0);
        c1 = WMMA_BF16(aq0, bk1a, c1);
        c1 = WMMA_BF16(aq1, bk1b, c1);

        float p0[8], p1[8], alpha[8];
#pragma unroll
        for (int e = 0; e < 8; ++e) {
            int   m    = hi * 8 + e;
            int   irow = it * 16 + m;
            int   j0   = jt * 32 + r;
            float s0 = (j0      <= irow) ? c0[e] * scale : -1e30f;
            float s1 = (j0 + 16 <= irow) ? c1[e] * scale : -1e30f;
            float mx = fmaxf(s0, s1);
            mx = fmaxf(mx, __shfl_xor(mx, 1));
            mx = fmaxf(mx, __shfl_xor(mx, 2));
            mx = fmaxf(mx, __shfl_xor(mx, 4));
            mx = fmaxf(mx, __shfl_xor(mx, 8));
            float mn = fmaxf(mi[e], mx);
            alpha[e] = __expf(mi[e] - mn);
            mi[e]    = mn;
            p0[e] = __expf(s0 - mn);
            p1[e] = __expf(s1 - mn);
            float rs = p0[e] + p1[e];
            rs += __shfl_xor(rs, 1);
            rs += __shfl_xor(rs, 2);
            rs += __shfl_xor(rs, 4);
            rs += __shfl_xor(rs, 8);
            li[e] = li[e] * alpha[e] + rs;
        }
#pragma unroll
        for (int dq = 0; dq < 4; ++dq)
#pragma unroll
            for (int e = 0; e < 8; ++e) oacc[dq][e] *= alpha[e];

        // P: C-layout -> A-layout via LDS (16x32 bf16, row-major)
        __syncthreads();
#pragma unroll
        for (int e = 0; e < 8; ++e) {
            int m = hi * 8 + e;
            Plds[m * 32 + r]      = (__bf16)p0[e];
            Plds[m * 32 + r + 16] = (__bf16)p1[e];
        }
        __syncthreads();
        v16bf pa;
        {
            union { v16bf v; uint4 q[2]; } u;
            u.q[0] = *(const uint4*)(&Plds[r * 32 + hi * 16]);
            u.q[1] = *(const uint4*)(&Plds[r * 32 + hi * 16 + 8]);
            pa = u.v;
        }
        const v16bf* vb = vbase + (size_t)jt * (4 * 32);
#pragma unroll
        for (int dq = 0; dq < 4; ++dq) {
            v16bf vv = vb[dq * 32 + lane];
            oacc[dq] = WMMA_BF16(pa, vv, oacc[dq]);
        }
    }

#pragma unroll
    for (int e = 0; e < 8; ++e) {
        int    m   = hi * 8 + e;
        float  inv = 1.0f / li[e];
        size_t row = ((size_t)b * SEQ + it * 16 + m) * D_MODEL + h * D_K;
#pragma unroll
        for (int dq = 0; dq < 4; ++dq)
            Ob[row + dq * 16 + r] = (__bf16)(oacc[dq][e] * inv);
    }
}

// ---------------------------------------------------------------------------
// Output projection: out = O(4096x1024 bf16) * Wo^T, fp32, TDM-staged Wo.
// grid = (64, 16), block 128.
// ---------------------------------------------------------------------------
__global__ void out_proj(const __bf16* __restrict__ Ob,
                         const __bf16* __restrict__ Wo,
                         float* __restrict__ out) {
    __shared__ __align__(64) __bf16 Blds[2][64 * 32];

    const int lane = threadIdx.x & 31;
    const int wave = __builtin_amdgcn_readfirstlane(threadIdx.x >> 5);
    const int r  = lane & 15;
    const int hi = lane >> 4;
    const int mt64 = blockIdx.x;
    const int nt   = blockIdx.y;

    const __bf16* wtile = Wo + (size_t)(nt * 64) * D_MODEL;
    const int     mt16  = mt64 * 4 + wave;
    const __bf16* arow  = Ob + (size_t)(mt16 * 16 + r) * D_MODEL;

    v8f c0, c1, c2, c3;
#pragma unroll
    for (int i = 0; i < 8; ++i) { c0[i] = 0.0f; c1[i] = 0.0f; c2[i] = 0.0f; c3[i] = 0.0f; }

    gemm64x64_tdm(arow, wtile, Blds, wave, r, hi, c0, c1, c2, c3);

    v8f cacc[4] = { c0, c1, c2, c3 };
#pragma unroll
    for (int dq = 0; dq < 4; ++dq)
#pragma unroll
        for (int e = 0; e < 8; ++e)
            out[(size_t)(mt16 * 16 + hi * 8 + e) * D_MODEL + nt * 64 + dq * 16 + r] = cacc[dq][e];
}

// ---------------------------------------------------------------------------
// Launch.  Workspace (40 MB):
//   [0,  8M)  xb (x bf16)  -- reused as Ob after qkv
//   [8M, 16M) wb: Wq,Wk,Wv,Wo bf16
//   [16M,24M) Qr   [24M,32M) Kr   [32M,40M) Vp
// ---------------------------------------------------------------------------
extern "C" void kernel_launch(void* const* d_in, const int* in_sizes, int n_in,
                              void* d_out, int out_size, void* d_ws, size_t ws_size,
                              hipStream_t stream) {
    const float* x  = (const float*)d_in[0];
    const float* Wq = (const float*)d_in[1];
    const float* Wk = (const float*)d_in[2];
    const float* Wv = (const float*)d_in[3];
    const float* Wo = (const float*)d_in[4];

    char* ws = (char*)d_ws;
    __bf16* xb = (__bf16*)(ws);
    __bf16* wb = (__bf16*)(ws + (8u  << 20));
    __bf16* Qr = (__bf16*)(ws + (16u << 20));
    __bf16* Kr = (__bf16*)(ws + (24u << 20));
    __bf16* Vp = (__bf16*)(ws + (32u << 20));
    __bf16* Ob = xb;

    const int NX = BATCH * SEQ * D_MODEL;
    const int NW = D_MODEL * D_MODEL;

    cvt_f32_to_bf16<<<NX / 256, 256, 0, stream>>>(x,  xb,          NX);
    cvt_f32_to_bf16<<<NW / 256, 256, 0, stream>>>(Wq, wb,          NW);
    cvt_f32_to_bf16<<<NW / 256, 256, 0, stream>>>(Wk, wb + 1 * NW, NW);
    cvt_f32_to_bf16<<<NW / 256, 256, 0, stream>>>(Wv, wb + 2 * NW, NW);
    cvt_f32_to_bf16<<<NW / 256, 256, 0, stream>>>(Wo, wb + 3 * NW, NW);

    qkv_gemm_rope<<<dim3(BATCH * SEQ / 64, NUM_HEADS, 3), 128, 0, stream>>>(
        xb, wb, Qr, Kr, Vp);

    attn_kernel<<<dim3(SEQ / 16, NUM_HEADS, BATCH), 32, 0, stream>>>(
        Qr, Kr, Vp, Ob);

    out_proj<<<dim3(BATCH * SEQ / 64, D_MODEL / 64), 128, 0, stream>>>(
        Ob, wb + 3 * (size_t)NW, (float*)d_out);
}
